// RNN_47356309405868
// MI455X (gfx1250) — matrix operand order repaired and checked by reading, not compiled, not run on hardware
//
#include <hip/hip_runtime.h>
#include <math.h>

// ---------------------------------------------------------------------------
// 2-layer tanh RNN for MI455X (gfx1250).  SEQ=2048, B=64, IN=H=512.
// bf16 WMMA (v_wmma_f32_16x16x32_bf16) with fp32 accumulate everywhere.
// Recurrent scan: batch-parallel (4 WGs x 16-row batch tiles, zero cross-WG
// sync), W_hh streamed from L2 in fragment-preshuffled order, proj tiles
// double-buffered into LDS via the Tensor Data Mover (tensor_load_to_lds +
// s_wait_tensorcnt), h kept as raw bf16 rows in LDS.
// ---------------------------------------------------------------------------

typedef __attribute__((ext_vector_type(16))) __bf16 v16bf;
typedef __attribute__((ext_vector_type(8)))  __bf16 v8bf;
typedef __attribute__((ext_vector_type(8)))  float  v8f;
typedef __attribute__((ext_vector_type(4)))  unsigned int u32x4;
typedef __attribute__((ext_vector_type(8)))  int          i32x8;
typedef __attribute__((ext_vector_type(4)))  int          i32x4;

#define SEQ_T   2048
#define BATCH_T 64

#if defined(__HIP_DEVICE_COMPILE__) && \
    __has_builtin(__builtin_amdgcn_tensor_load_to_lds)
#define HAVE_TDM 1
#else
#define HAVE_TDM 0
#endif

union AFrag { v16bf v; v8bf h[2]; };

__device__ __forceinline__ unsigned short f32_to_bf16(float f) {
  unsigned int u = __float_as_uint(f);
  unsigned int r = u + 0x7FFFu + ((u >> 16) & 1u);   // round-to-nearest-even
  return (unsigned short)(r >> 16);
}

__device__ __forceinline__ v8f wmma_bf16(v16bf a, v16bf b, v8f c) {
  return __builtin_amdgcn_wmma_f32_16x16x32_bf16(
      false, a, false, b, (short)0, c, false, false);
}

// ---------------------------------------------------------------------------
// Pre-shuffle a [512 out, 512 in] fp32 weight matrix (row-major W[n,k]) into
// bf16 WMMA B-fragment order:
//   frag[(nt*8192) + ks*512 + lane*16 + e],  n = nt*16 + (lane&15),
//   k = ks*32 + (lane>=16 ? 8 : 0) + (e&7) + (e&8 ? 16 : 0)   (B lane: table K order)
// => each lane's 32-byte fragment is contiguous; kstep stride is a 1 KB
//    immediate offset in the consumer loops.
// ---------------------------------------------------------------------------
__global__ void prep_wfrag(const float* __restrict__ W,
                           unsigned short* __restrict__ out) {
  int i = blockIdx.x * blockDim.x + threadIdx.x;   // 262144 elements
  int e    = i & 15;
  int lane = (i >> 4) & 31;
  int ks   = (i >> 9) & 15;
  int nt   = i >> 13;
  int k = ks * 32 + ((lane >= 16) ? 8 : 0) + (e & 7) + ((e & 8) ? 16 : 0);
  int n = nt * 16 + (lane & 15);
  out[i] = f32_to_bf16(W[n * 512 + k]);
}

// ---------------------------------------------------------------------------
// Tensor Data Mover: async copy of one contiguous 32 KB fp32 tile into LDS.
// 1-D tensor descriptor: data_size=4B, tensor_dim0=tile_dim0=8192, tile_dim1=1.
// Issued once per workgroup (wave 0); completion via s_wait_tensorcnt.
// 6-arg builtin form (clang-23 / therock headers): g0, g1, g2, g3, g4, cpol.
// Fallback (builtin absent): cooperative vector copy.
// ---------------------------------------------------------------------------
__device__ __forceinline__ void tile_prefetch(const float* __restrict__ src,
                                              float* __restrict__ lds_dst,
                                              int tid) {
#if HAVE_TDM
  if (tid < 32) {
    u32x4 g0;
    unsigned long long ga = (unsigned long long)(size_t)src;
    g0[0] = 1u;                                           // count=1
    g0[1] = (unsigned)(size_t)(void*)lds_dst;             // LDS byte offset
    g0[2] = (unsigned)ga;                                 // global_addr[31:0]
    g0[3] = (unsigned)((ga >> 32) & 0x1FFFFFFull) | (2u << 30); // addr+type=2
    i32x8 g1;
    g1[0] = 0x20000;              // wg_mask=0, data_size=2 (4 bytes)
    g1[1] = (int)(8192u << 16);   // tensor_dim0[15:0] in [31:16]
    g1[2] = (int)(1u << 16);      // tensor_dim0 hi=0; tensor_dim1[15:0]=1
    g1[3] = (int)(8192u << 16);   // tensor_dim1 hi=0; tile_dim0=8192
    g1[4] = 1;                    // tile_dim1=1, tile_dim2=0
    g1[5] = 8192;                 // tensor_dim0_stride[31:0]
    g1[6] = 0;
    g1[7] = 0;
    i32x4 z4 = {0, 0, 0, 0};
    i32x8 z8 = {0, 0, 0, 0, 0, 0, 0, 0};
    __builtin_amdgcn_tensor_load_to_lds(g0, g1, z4, z4, z8, 0);
  }
#else
  const float4* s4 = (const float4*)src;
  float4* d4 = (float4*)lds_dst;
  for (int i = tid; i < 2048; i += 256) d4[i] = s4[i];
#endif
}

__device__ __forceinline__ void tile_wait(int tid) {
#if HAVE_TDM
  if (tid < 32) {
#if __has_builtin(__builtin_amdgcn_s_wait_tensorcnt)
    __builtin_amdgcn_s_wait_tensorcnt(0);
#endif
  }
#endif
}

// ---------------------------------------------------------------------------
// Pipelined 16-kstep x 4-ntile WMMA block (B fragments double-buffered,
// immediate-offset global loads).  A fragment source differs per kernel, so
// the loop body is expressed via a macro over an A-load expression.
// ---------------------------------------------------------------------------
#define WMMA_KLOOP(ALOAD0, ALOAD1)                                          \
  {                                                                         \
    v16bf b0 = *(const v16bf*)(wb0);                                        \
    v16bf b1 = *(const v16bf*)(wb1);                                        \
    v16bf b2 = *(const v16bf*)(wb2);                                        \
    v16bf b3 = *(const v16bf*)(wb3);                                        \
    _Pragma("unroll")                                                       \
    for (int ks = 0; ks < 16; ++ks) {                                       \
      AFrag a;                                                              \
      a.h[0] = (ALOAD0);                                                    \
      a.h[1] = (ALOAD1);                                                    \
      v16bf n0, n1, n2, n3;                                                 \
      if (ks < 15) {                                                        \
        n0 = *(const v16bf*)(wb0 + (ks + 1) * 512);                         \
        n1 = *(const v16bf*)(wb1 + (ks + 1) * 512);                         \
        n2 = *(const v16bf*)(wb2 + (ks + 1) * 512);                         \
        n3 = *(const v16bf*)(wb3 + (ks + 1) * 512);                         \
      }                                                                     \
      acc[0] = wmma_bf16(a.v, b0, acc[0]);                                  \
      acc[1] = wmma_bf16(a.v, b1, acc[1]);                                  \
      acc[2] = wmma_bf16(a.v, b2, acc[2]);                                  \
      acc[3] = wmma_bf16(a.v, b3, acc[3]);                                  \
      if (ks < 15) { b0 = n0; b1 = n1; b2 = n2; b3 = n3; }                  \
    }                                                                       \
  }

// ---------------------------------------------------------------------------
// Projection GEMM (layer 0): C[M,512] = X_f32[M,512] . W^T + bias.
// One WG per 16-row M tile (tile = 32 KB contiguous); X staged once to LDS
// as raw bf16 rows (X read exactly once from HBM).
// ---------------------------------------------------------------------------
__global__ __launch_bounds__(256) void proj_gemm_f32(
    const float* __restrict__ X, const unsigned short* __restrict__ WF,
    const float* __restrict__ bias, float* __restrict__ C) {
  __shared__ unsigned short xbf[16 * 512];   // 16 KB bf16 tile rows

  const int tid = threadIdx.x;
  const int lane = tid & 31, wv = tid >> 5, lm = lane & 15, lhi = lane >> 4;
  const int mrow = blockIdx.x * 16;

  {  // stage: 8192 contiguous floats -> bf16 rows (coalesced float4 loads)
    const float4* s4 = (const float4*)(X + (size_t)mrow * 512);
    uint2* d2 = (uint2*)xbf;
    for (int i = tid; i < 2048; i += 256) {
      float4 f = s4[i];
      uint2 p;
      p.x = (unsigned)f32_to_bf16(f.x) | ((unsigned)f32_to_bf16(f.y) << 16);
      p.y = (unsigned)f32_to_bf16(f.z) | ((unsigned)f32_to_bf16(f.w) << 16);
      d2[i] = p;
    }
  }
  __syncthreads();

  const unsigned short* wb0 = WF + (size_t)(wv * 4 + 0) * 8192 + lane * 16;
  const unsigned short* wb1 = WF + (size_t)(wv * 4 + 1) * 8192 + lane * 16;
  const unsigned short* wb2 = WF + (size_t)(wv * 4 + 2) * 8192 + lane * 16;
  const unsigned short* wb3 = WF + (size_t)(wv * 4 + 3) * 8192 + lane * 16;
  const unsigned short* hrow = &xbf[lm * 512];

  v8f acc[4];
#pragma unroll
  for (int j = 0; j < 4; ++j)
    acc[j] = (v8f){0.f, 0.f, 0.f, 0.f, 0.f, 0.f, 0.f, 0.f};

  WMMA_KLOOP(*(const v8bf*)&hrow[ks * 32 + lhi * 8],
             *(const v8bf*)&hrow[ks * 32 + lhi * 8 + 16])

#pragma unroll
  for (int j = 0; j < 4; ++j) {
    int n = (wv * 4 + j) * 16 + lm;
    float bb = bias[n];
#pragma unroll
    for (int r = 0; r < 8; ++r)
      C[(size_t)(mrow + r + lhi * 8) * 512 + n] = acc[j][r] + bb;
  }
}

// ---------------------------------------------------------------------------
// Projection GEMM (layer 1): input already stored as raw bf16 rows by scan0;
// A fragments loaded straight from global (L2-resident per M tile).
// ---------------------------------------------------------------------------
__global__ __launch_bounds__(256) void proj_gemm_bf16(
    const unsigned short* __restrict__ XB, const unsigned short* __restrict__ WF,
    const float* __restrict__ bias, float* __restrict__ C) {
  const int tid = threadIdx.x;
  const int lane = tid & 31, wv = tid >> 5, lm = lane & 15, lhi = lane >> 4;
  const int mrow = blockIdx.x * 16;

  const unsigned short* wb0 = WF + (size_t)(wv * 4 + 0) * 8192 + lane * 16;
  const unsigned short* wb1 = WF + (size_t)(wv * 4 + 1) * 8192 + lane * 16;
  const unsigned short* wb2 = WF + (size_t)(wv * 4 + 2) * 8192 + lane * 16;
  const unsigned short* wb3 = WF + (size_t)(wv * 4 + 3) * 8192 + lane * 16;
  const unsigned short* arow = XB + (size_t)(mrow + lm) * 512 + lhi * 8;

  v8f acc[4];
#pragma unroll
  for (int j = 0; j < 4; ++j)
    acc[j] = (v8f){0.f, 0.f, 0.f, 0.f, 0.f, 0.f, 0.f, 0.f};

  WMMA_KLOOP(*(const v8bf*)(arow + ks * 32),
             *(const v8bf*)(arow + ks * 32 + 16))

#pragma unroll
  for (int j = 0; j < 4; ++j) {
    int n = (wv * 4 + j) * 16 + lm;
    float bb = bias[n];
#pragma unroll
    for (int r = 0; r < 8; ++r)
      C[(size_t)(mrow + r + lhi * 8) * 512 + n] = acc[j][r] + bb;
  }
}

// ---------------------------------------------------------------------------
// Recurrent scan:  h_t = tanh(proj[t] + h_{t-1} . W_hh^T + b_hh)
// 4 WGs, each owning a 16-row batch tile (independent => no cross-WG sync).
// h kept as raw bf16 rows in LDS; proj tiles (32 KB, contiguous) double-
// buffered via TDM one step ahead; W_hh fragments stream from L2.
// write_f32 != 0 -> emit fp32 out (layer 1); obf != null -> emit bf16 rows
// for the next layer's projection (layer 0).
// ---------------------------------------------------------------------------
__global__ __launch_bounds__(256) void rnn_scan(
    const float* __restrict__ proj, const unsigned short* __restrict__ WF,
    const float* __restrict__ bhh, float* __restrict__ out,
    unsigned short* __restrict__ obf, float* __restrict__ hidden,
    int write_f32) {
  __shared__ unsigned short hbf[16 * 512];   // 16 KB: h as raw bf16 rows
  __shared__ float pshm[2][16 * 512];        // 2 x 32 KB proj tile buffers

  const int tid = threadIdx.x;
  const int lane = tid & 31, wv = tid >> 5, lm = lane & 15, lhi = lane >> 4;
  const int brow0 = blockIdx.x * 16;

  for (int i = tid; i < 16 * 512; i += 256) hbf[i] = 0;   // h_{-1} = 0

  tile_prefetch(proj + (size_t)brow0 * 512, pshm[0], tid);  // tile t=0

  float bb[4];
#pragma unroll
  for (int j = 0; j < 4; ++j) bb[j] = bhh[(wv * 4 + j) * 16 + lm];

  const unsigned short* wb0 = WF + (size_t)(wv * 4 + 0) * 8192 + lane * 16;
  const unsigned short* wb1 = WF + (size_t)(wv * 4 + 1) * 8192 + lane * 16;
  const unsigned short* wb2 = WF + (size_t)(wv * 4 + 2) * 8192 + lane * 16;
  const unsigned short* wb3 = WF + (size_t)(wv * 4 + 3) * 8192 + lane * 16;
  const unsigned short* hrow = &hbf[lm * 512];

  for (int t = 0; t < SEQ_T; ++t) {
    tile_wait(tid);
    __syncthreads();   // proj tile t arrived; h_{t-1} rows visible

    if (t + 1 < SEQ_T)  // overlap next tile's DMA with this step's compute
      tile_prefetch(proj + (size_t)((t + 1) * BATCH_T + brow0) * 512,
                    pshm[(t + 1) & 1], tid);

    const float* pt = pshm[t & 1];
    v8f acc[4];
#pragma unroll
    for (int j = 0; j < 4; ++j) {
      int n = (wv * 4 + j) * 16 + lm;
#pragma unroll
      for (int r = 0; r < 8; ++r)
        acc[j][r] = pt[(r + lhi * 8) * 512 + n];
    }

    WMMA_KLOOP(*(const v8bf*)&hrow[ks * 32 + lhi * 8],
               *(const v8bf*)&hrow[ks * 32 + lhi * 8 + 16])

    __syncthreads();   // everyone done reading h_{t-1}

    const size_t prow = (size_t)t * BATCH_T + brow0;
#pragma unroll
    for (int j = 0; j < 4; ++j) {
      int n = (wv * 4 + j) * 16 + lm;
#pragma unroll
      for (int r = 0; r < 8; ++r) {
        int m = r + lhi * 8;
        float v = tanhf(acc[j][r] + bb[j]);
        unsigned short vb = f32_to_bf16(v);
        hbf[m * 512 + n] = vb;
        if (write_f32) out[(prow + m) * 512 + n] = v;
        if (obf)       obf[(prow + m) * 512 + n] = vb;
        if (t == SEQ_T - 1) hidden[(size_t)(brow0 + m) * 512 + n] = v;
      }
    }
  }
}

// ---------------------------------------------------------------------------
// prep -> proj0(f32 x) -> scan0(bf16 out0 -> ws) -> proj1(bf16) -> scan1(f32
// out1 -> d_out, hidden states appended).
// ---------------------------------------------------------------------------
extern "C" void kernel_launch(void* const* d_in, const int* in_sizes, int n_in,
                              void* d_out, int out_size, void* d_ws,
                              size_t ws_size, hipStream_t stream) {
  (void)in_sizes; (void)n_in; (void)out_size; (void)ws_size;

  const float* x     = (const float*)d_in[0];
  const float* W_ih0 = (const float*)d_in[1];
  const float* W_hh0 = (const float*)d_in[2];
  const float* b_ih0 = (const float*)d_in[3];
  const float* b_hh0 = (const float*)d_in[4];
  const float* W_ih1 = (const float*)d_in[5];
  const float* W_hh1 = (const float*)d_in[6];
  const float* b_ih1 = (const float*)d_in[7];
  const float* b_hh1 = (const float*)d_in[8];

  const size_t WELEM = 512 * 512;                 // per weight matrix
  const size_t M     = (size_t)SEQ_T * BATCH_T;   // 131072 rows
  unsigned short* wf_ih0 = (unsigned short*)d_ws;
  unsigned short* wf_hh0 = wf_ih0 + WELEM;
  unsigned short* wf_ih1 = wf_hh0 + WELEM;
  unsigned short* wf_hh1 = wf_ih1 + WELEM;
  float* projbuf = (float*)((char*)d_ws + 4 * WELEM * sizeof(unsigned short));
  unsigned short* obf = (unsigned short*)(projbuf + M * 512);  // bf16 out0

  float* out  = (float*)d_out;
  float* hid0 = out + M * 512;                    // hidden tail [2,64,512]
  float* hid1 = hid0 + (size_t)BATCH_T * 512;

  prep_wfrag<<<1024, 256, 0, stream>>>(W_ih0, wf_ih0);
  prep_wfrag<<<1024, 256, 0, stream>>>(W_hh0, wf_hh0);
  prep_wfrag<<<1024, 256, 0, stream>>>(W_ih1, wf_ih1);
  prep_wfrag<<<1024, 256, 0, stream>>>(W_hh1, wf_hh1);

  proj_gemm_f32<<<8192, 256, 0, stream>>>(x, wf_ih0, b_ih0, projbuf);
  rnn_scan<<<4, 256, 0, stream>>>(projbuf, wf_hh0, b_hh0, out, obf, hid0, 0);
  proj_gemm_bf16<<<8192, 256, 0, stream>>>(obf, wf_ih1, b_ih1, projbuf);
  rnn_scan<<<4, 256, 0, stream>>>(projbuf, wf_hh1, b_hh1, out, (unsigned short*)nullptr, hid1, 1);
}
